// RelativePosMultiheadAttention_6854767804523
// MI455X (gfx1250) — matrix operand order repaired and checked by reading, not compiled, not run on hardware
//
#include <hip/hip_runtime.h>
#include <math.h>

// RelativePosMultiheadAttention for MI455X (gfx1250, wave32, WMMA f16->f32).
// B=2, T=2048, D=1024, H=16, HD=64.
//
// Pipeline (all on `stream`):
//   1. cvt_f16:      x (f32) -> xh (f16)
//   2. transpose16:  Wqkv -> WqkvT (f16, [3072][1024]), Wout -> WoutT (f16)
//   3. build_ehat:   Ehat[h][d][hd] = PE(distance d, col h*64+hd), f16
//   4. qkv_gemm:     qh/kh row-major [bh][t][hd] f16, v transposed vt [bh][hd][t] f16
//   5. attn:         two-pass softmax per 16-row q tile; writes normalized score
//                    (537MB, exactly once) + ctxh f16 [b*t][d]
//   6. out_gemm:     out = ctxh @ Wout + bout (f32)

#define T_ 2048
#define D_ 1024
#define B_ 2
#define H_ 16
#define HD_ 64
#define BH_ (B_ * H_)
#define BT_ (B_ * T_)

#define QE_LD 34   // float stride: halves of the wave hit disjoint bank groups
#define PL_LD 40   // f16 stride: b128 A-reloads spread over all 64 banks

typedef _Float16 v16h __attribute__((ext_vector_type(16)));
typedef _Float16 h8   __attribute__((ext_vector_type(8)));
typedef float    v8f  __attribute__((ext_vector_type(8)));

__device__ __forceinline__ v8f wmma32(v16h a, v16h b, v8f c) {
  // D = A(16x32 f16) * B(32x16 f16) + C(16x16 f32)
  return __builtin_amdgcn_wmma_f32_16x16x32_f16(false, a, false, b, (short)0, c,
                                                false, false);
}

// A-matrix (16x32 f16) loader from row-major f16 (row stride ld halves).
// Layout: lanes 0-15 row M=lane, K = 0..15; lanes 16-31 same row, K = 16..31
// (matches load_b16).  -> two contiguous 16B loads/lane.
__device__ __forceinline__ v16h load_a16(const _Float16* p, int ld) {
  const int lane = threadIdx.x & 31;
  const int m = lane & 15;
  const int kb = (lane >> 4) * 16;
  const h8 lo = *(const h8*)(p + m * ld + kb);
  const h8 hi = *(const h8*)(p + m * ld + 8 + kb);
  v16h a;
#pragma unroll
  for (int i = 0; i < 8; ++i) { a[i] = lo[i]; a[8 + i] = hi[i]; }
  return a;
}

// B-matrix (32x16 f16) loader where B[k][n] = src[n*ld + k] (src holds the 16
// columns as contiguous rows). Lanes 0-15: col n=lane, K=0..15; lanes 16-31:
// K=16..31.  -> one contiguous 32B load per lane.
__device__ __forceinline__ v16h load_b16(const _Float16* p, int ld) {
  const int lane = threadIdx.x & 31;
  const int n = lane & 15;
  const int kb = (lane >> 4) * 16;
  return *(const v16h*)(p + n * ld + kb);
}

// ---------------------------------------------------------------- helpers ---
__global__ void cvt_f16(const float* __restrict__ src, _Float16* __restrict__ dst,
                        int n) {
  int i = blockIdx.x * blockDim.x + threadIdx.x;
  if (i < n) dst[i] = (_Float16)src[i];
}

// dst[c*rows + r] = (f16) src[r*cols + c]
__global__ void transpose16(const float* __restrict__ src,
                            _Float16* __restrict__ dst, int rows, int cols) {
  int i = blockIdx.x * blockDim.x + threadIdx.x;
  if (i >= rows * cols) return;
  int r = i / cols, c = i - r * cols;
  dst[(size_t)c * rows + r] = (_Float16)src[i];
}

// Ehat[h][d][hd]: sinusoid at relative distance d, model column h*64+hd.
// (Reference E row l = T-1-d has pos=d; skew gives att_e[i,j] = q_i . PE(i-j).)
__global__ void build_ehat(_Float16* __restrict__ ehat) {
  int i = blockIdx.x * blockDim.x + threadIdx.x;
  if (i >= H_ * T_ * HD_) return;
  int hd = i & 63;
  int d = (i >> 6) & (T_ - 1);
  int h = i >> 17;
  int col = h * HD_ + hd;
  float div = expf(-logf(10000.f) * (float)(col & ~1) / (float)D_);
  float ang = (float)d * div;
  ehat[i] = (_Float16)((col & 1) ? cosf(ang) : sinf(ang));
}

// ------------------------------------------------------------- qkv GEMM -----
// [BT, D] x [D, 3D]; block = 8 waves, each wave a 32x64 tile (two 16-row A
// strips sharing 4 B tiles -> 8 independent WMMAs per K-chunk). Next chunk's
// A+B tiles are loaded before the current chunk's WMMAs (double buffered).
__global__ __launch_bounds__(256) void qkv_gemm(
    const _Float16* __restrict__ xh, const _Float16* __restrict__ wT,
    _Float16* __restrict__ qh, _Float16* __restrict__ kh,
    _Float16* __restrict__ vt) {
  const int wave = threadIdx.x >> 5, lane = threadIdx.x & 31;
  const int c0 = blockIdx.x * 64;
  const int r0 = blockIdx.y * 256 + wave * 32;
  const int hi = lane >> 4, N = lane & 15;

  const _Float16* aP0 = xh + (size_t)r0 * D_;
  const _Float16* aP1 = xh + (size_t)(r0 + 16) * D_;
  const _Float16* bP  = wT + (size_t)c0 * D_;

  v16h a0 = load_a16(aP0, D_);
  v16h a1 = load_a16(aP1, D_);
  v16h b0 = load_b16(bP, D_);
  v16h b1 = load_b16(bP + 16 * D_, D_);
  v16h b2 = load_b16(bP + 32 * D_, D_);
  v16h b3 = load_b16(bP + 48 * D_, D_);
  v8f acc0[4] = {}, acc1[4] = {};

  for (int kc = 0; kc < D_; kc += 32) {
    const int kn = (kc + 32 < D_) ? kc + 32 : 0;  // last iter: dummy reload
    v16h an0 = load_a16(aP0 + kn, D_);
    v16h an1 = load_a16(aP1 + kn, D_);
    v16h bn0 = load_b16(bP + kn, D_);
    v16h bn1 = load_b16(bP + 16 * D_ + kn, D_);
    v16h bn2 = load_b16(bP + 32 * D_ + kn, D_);
    v16h bn3 = load_b16(bP + 48 * D_ + kn, D_);
    acc0[0] = wmma32(a0, b0, acc0[0]);
    acc1[0] = wmma32(a1, b0, acc1[0]);
    acc0[1] = wmma32(a0, b1, acc0[1]);
    acc1[1] = wmma32(a1, b1, acc1[1]);
    acc0[2] = wmma32(a0, b2, acc0[2]);
    acc1[2] = wmma32(a1, b2, acc1[2]);
    acc0[3] = wmma32(a0, b3, acc0[3]);
    acc1[3] = wmma32(a1, b3, acc1[3]);
    a0 = an0; a1 = an1; b0 = bn0; b1 = bn1; b2 = bn2; b3 = bn3;
  }

#pragma unroll
  for (int s = 0; s < 2; ++s) {
    const v8f* acc = s ? acc1 : acc0;
    const int rr = r0 + s * 16;
    const int b = rr >> 11;          // tiles never straddle the batch boundary
    const int t0 = rr & (T_ - 1);
#pragma unroll
    for (int g = 0; g < 4; ++g) {
      int c = c0 + g * 16 + N;
      int which = c >> 10;           // 0=q 1=k 2=v
      int dcol = c & (D_ - 1);
      int h = dcol >> 6, hd = dcol & 63;
      int bh = b * H_ + h;
      if (which == 2) {              // v transposed: [bh][hd][t], pack 8 t's
        h8 pk;
#pragma unroll
        for (int r = 0; r < 8; ++r) pk[r] = (_Float16)acc[g][r];
        *(h8*)(vt + ((size_t)bh * HD_ + hd) * T_ + t0 + hi * 8) = pk;
      } else {
        _Float16* dst = (which == 0) ? qh : kh;
#pragma unroll
        for (int r = 0; r < 8; ++r) {
          int t = t0 + r + hi * 8;
          dst[((size_t)bh * T_ + t) * HD_ + hd] = (_Float16)acc[g][r];
        }
      }
    }
  }
}

// ------------------------------------------------------------ attention -----
// One wave per block; 16 q rows per wave; 32-key chunks; two passes over K
// (stats pass, then emit pass). Operands are L2-resident so recompute is cheap
// and the 537MB score tensor is written exactly once, already normalized.
__global__ __launch_bounds__(32) void attn(
    const _Float16* __restrict__ qh, const _Float16* __restrict__ kh,
    const _Float16* __restrict__ vt, const _Float16* __restrict__ ehat,
    float* __restrict__ score, _Float16* __restrict__ ctxh) {
  const int bh = blockIdx.y;
  const int h = bh & (H_ - 1);
  const int b = bh >> 4;
  const int i0 = blockIdx.x * 16;
  const int lane = threadIdx.x & 31;
  const int hi = lane >> 4, N = lane & 15;

  __shared__ __align__(16) float    qe[16 * QE_LD];
  __shared__ __align__(16) _Float16 pl[16 * PL_LD];

  const _Float16* qbase = qh + ((size_t)bh * T_ + i0) * HD_;
  const _Float16* kbase = kh + (size_t)bh * T_ * HD_;
  const _Float16* vbase = vt + (size_t)bh * HD_ * T_;
  const _Float16* ebase = ehat + (size_t)h * T_ * HD_;

  const v16h aq0 = load_a16(qbase, HD_);       // hd 0..31
  const v16h aq1 = load_a16(qbase + 32, HD_);  // hd 32..63

  // logits for one 16x16 subtile at key offset js0 (this lane's 8 elements);
  // all 6 B tiles are loaded up-front so loads overlap the 6 WMMAs.
  auto computeS = [&](int js0, float sOut[8]) {
    const _Float16* kp = kbase + (size_t)js0 * HD_;
    const int dbase = i0 - js0 - 15;   // rel window: w = (i-i0)-(j-js0)+15
    const int kb = hi * 16;
    int r0e = dbase + N;      r0e = r0e < 0 ? 0 : (r0e > T_ - 1 ? T_ - 1 : r0e);
    int r1e = dbase + 16 + N; r1e = r1e < 0 ? 0 : (r1e > T_ - 1 ? T_ - 1 : r1e);
    const v16h bk0  = load_b16(kp, HD_);
    const v16h bk1  = load_b16(kp + 32, HD_);
    const v16h be00 = *(const v16h*)(ebase + (size_t)r0e * HD_ + kb);
    const v16h be01 = *(const v16h*)(ebase + (size_t)r0e * HD_ + 32 + kb);
    const v16h be10 = *(const v16h*)(ebase + (size_t)r1e * HD_ + kb);
    const v16h be11 = *(const v16h*)(ebase + (size_t)r1e * HD_ + 32 + kb);
    v8f ak = {}, ae0 = {}, ae1 = {};
    ak  = wmma32(aq0, bk0, ak);
    ae0 = wmma32(aq0, be00, ae0);
    ae1 = wmma32(aq0, be10, ae1);
    ak  = wmma32(aq1, bk1, ak);
    ae0 = wmma32(aq1, be01, ae0);
    ae1 = wmma32(aq1, be11, ae1);
    __syncthreads();
#pragma unroll
    for (int r = 0; r < 8; ++r) {
      int M = r + hi * 8;
      qe[M * QE_LD + N] = ae0[r];
      qe[M * QE_LD + 16 + N] = ae1[r];
    }
    __syncthreads();
#pragma unroll
    for (int r = 0; r < 8; ++r) {
      int M = r + hi * 8;
      float e = qe[M * QE_LD + (M - N + 15)];    // lane-consecutive reads
      float v = (ak[r] + e) * 0.125f;            // 1/sqrt(64)
      sOut[r] = ((js0 + N) > (i0 + M)) ? -INFINITY : v;
    }
  };

  float m8[8], l8[8];
#pragma unroll
  for (int r = 0; r < 8; ++r) { m8[r] = -INFINITY; l8[r] = 0.f; }

  const int jstop = i0 + 16;
  // ----- pass A: row max + denominator -----
  for (int j0 = 0; j0 < jstop; j0 += 32) {
    __builtin_prefetch(kbase + (size_t)(j0 + 32) * HD_, 0, 1);
    float s0[8], s1[8];
    computeS(j0, s0);
    computeS(j0 + 16, s1);
#pragma unroll
    for (int r = 0; r < 8; ++r) {
      float mx = fmaxf(s0[r], s1[r]);
      mx = fmaxf(mx, __shfl_xor(mx, 1));
      mx = fmaxf(mx, __shfl_xor(mx, 2));
      mx = fmaxf(mx, __shfl_xor(mx, 4));
      mx = fmaxf(mx, __shfl_xor(mx, 8));
      float mn = fmaxf(m8[r], mx);
      float sm = expf(s0[r] - mn) + expf(s1[r] - mn);
      sm += __shfl_xor(sm, 1);
      sm += __shfl_xor(sm, 2);
      sm += __shfl_xor(sm, 4);
      sm += __shfl_xor(sm, 8);
      l8[r] = l8[r] * expf(m8[r] - mn) + sm;
      m8[r] = mn;
    }
  }
  float rl8[8];
#pragma unroll
  for (int r = 0; r < 8; ++r) rl8[r] = 1.0f / l8[r];

  // ----- pass B: emit normalized score + accumulate ctx = P @ V -----
  v8f ctx[4] = {};
  float* srow = score + (size_t)bh * T_ * T_;
  for (int j0 = 0; j0 < jstop; j0 += 32) {
    float s0[8], s1[8];
    computeS(j0, s0);
    computeS(j0 + 16, s1);
#pragma unroll
    for (int r = 0; r < 8; ++r) {
      int M = r + hi * 8;
      float p0 = expf(s0[r] - m8[r]) * rl8[r];   // masked -> exp(-inf)=0
      float p1 = expf(s1[r] - m8[r]) * rl8[r];
      srow[(size_t)(i0 + M) * T_ + j0 + N] = p0;
      srow[(size_t)(i0 + M) * T_ + j0 + 16 + N] = p1;
      pl[M * PL_LD + N] = (_Float16)p0;
      pl[M * PL_LD + 16 + N] = (_Float16)p1;
    }
    __syncthreads();
    v16h ap = load_a16((const _Float16*)pl, PL_LD);  // P tile, A layout (K=32)
    __syncthreads();
    const v16h bv0 = *(const v16h*)(vbase + (size_t)(0 * 16 + N) * T_ + j0 + hi * 16);
    const v16h bv1 = *(const v16h*)(vbase + (size_t)(1 * 16 + N) * T_ + j0 + hi * 16);
    const v16h bv2 = *(const v16h*)(vbase + (size_t)(2 * 16 + N) * T_ + j0 + hi * 16);
    const v16h bv3 = *(const v16h*)(vbase + (size_t)(3 * 16 + N) * T_ + j0 + hi * 16);
    ctx[0] = wmma32(ap, bv0, ctx[0]);
    ctx[1] = wmma32(ap, bv1, ctx[1]);
    ctx[2] = wmma32(ap, bv2, ctx[2]);
    ctx[3] = wmma32(ap, bv3, ctx[3]);
  }

  // store ctx (f16, [b*t][d] row-major for the output GEMM)
#pragma unroll
  for (int g = 0; g < 4; ++g) {
#pragma unroll
    for (int r = 0; r < 8; ++r) {
      int M = r + hi * 8;
      ctxh[((size_t)(b * T_ + i0 + M)) * D_ + h * HD_ + g * 16 + N] =
          (_Float16)ctx[g][r];
    }
  }

  // zero-fill the never-touched upper-triangle chunks of score
  const int jend = (jstop + 31) & ~31;
  for (int M = 0; M < 16; ++M) {
    float* row = srow + (size_t)(i0 + M) * T_;
    for (int jj = jend + lane * 4; jj < T_; jj += 128)
      *(float4*)(row + jj) = make_float4(0.f, 0.f, 0.f, 0.f);
  }
}

// ------------------------------------------------------------- out GEMM -----
// Same 32x64-per-wave double-buffered scheme as qkv_gemm.
__global__ __launch_bounds__(256) void out_gemm(
    const _Float16* __restrict__ ctxh, const _Float16* __restrict__ woT,
    const float* __restrict__ bout, float* __restrict__ out) {
  const int wave = threadIdx.x >> 5, lane = threadIdx.x & 31;
  const int c0 = blockIdx.x * 64;
  const int r0 = blockIdx.y * 256 + wave * 32;
  const int hi = lane >> 4, N = lane & 15;

  const _Float16* aP0 = ctxh + (size_t)r0 * D_;
  const _Float16* aP1 = ctxh + (size_t)(r0 + 16) * D_;
  const _Float16* bP  = woT + (size_t)c0 * D_;

  v16h a0 = load_a16(aP0, D_);
  v16h a1 = load_a16(aP1, D_);
  v16h b0 = load_b16(bP, D_);
  v16h b1 = load_b16(bP + 16 * D_, D_);
  v16h b2 = load_b16(bP + 32 * D_, D_);
  v16h b3 = load_b16(bP + 48 * D_, D_);
  v8f acc0[4] = {}, acc1[4] = {};

  for (int kc = 0; kc < D_; kc += 32) {
    const int kn = (kc + 32 < D_) ? kc + 32 : 0;
    v16h an0 = load_a16(aP0 + kn, D_);
    v16h an1 = load_a16(aP1 + kn, D_);
    v16h bn0 = load_b16(bP + kn, D_);
    v16h bn1 = load_b16(bP + 16 * D_ + kn, D_);
    v16h bn2 = load_b16(bP + 32 * D_ + kn, D_);
    v16h bn3 = load_b16(bP + 48 * D_ + kn, D_);
    acc0[0] = wmma32(a0, b0, acc0[0]);
    acc1[0] = wmma32(a1, b0, acc1[0]);
    acc0[1] = wmma32(a0, b1, acc0[1]);
    acc1[1] = wmma32(a1, b1, acc1[1]);
    acc0[2] = wmma32(a0, b2, acc0[2]);
    acc1[2] = wmma32(a1, b2, acc1[2]);
    acc0[3] = wmma32(a0, b3, acc0[3]);
    acc1[3] = wmma32(a1, b3, acc1[3]);
    a0 = an0; a1 = an1; b0 = bn0; b1 = bn1; b2 = bn2; b3 = bn3;
  }

#pragma unroll
  for (int s = 0; s < 2; ++s) {
    const v8f* acc = s ? acc1 : acc0;
    const int rr = r0 + s * 16;
#pragma unroll
    for (int g = 0; g < 4; ++g) {
      int c = c0 + g * 16 + N;
      float bias = bout[c];
#pragma unroll
      for (int r = 0; r < 8; ++r)
        out[(size_t)(rr + r + hi * 8) * D_ + c] = acc[g][r] + bias;
    }
  }
}

// ---------------------------------------------------------------------------
extern "C" void kernel_launch(void* const* d_in, const int* in_sizes, int n_in,
                              void* d_out, int out_size, void* d_ws,
                              size_t ws_size, hipStream_t stream) {
  const float* x    = (const float*)d_in[0];
  // d_in[1] = mask (causal, known analytically -> unused)
  const float* Wqkv = (const float*)d_in[2];
  const float* Wout = (const float*)d_in[3];
  const float* bout = (const float*)d_in[4];

  float* out   = (float*)d_out;
  float* score = (float*)d_out + (size_t)BT_ * D_;

  // workspace carve-up (f16 staging, ~52 MB total)
  char* ws = (char*)d_ws;
  size_t off = 0;
  auto carve = [&](size_t bytes) {
    void* p = ws + off;
    off += (bytes + 255) & ~(size_t)255;
    return p;
  };
  _Float16* xh    = (_Float16*)carve((size_t)BT_ * D_ * 2);       // 8 MB
  _Float16* wqkvT = (_Float16*)carve((size_t)3 * D_ * D_ * 2);    // 6 MB
  _Float16* woutT = (_Float16*)carve((size_t)D_ * D_ * 2);        // 2 MB
  _Float16* qh    = (_Float16*)carve((size_t)BH_ * T_ * HD_ * 2); // 8 MB
  _Float16* kh    = (_Float16*)carve((size_t)BH_ * T_ * HD_ * 2); // 8 MB
  _Float16* vt    = (_Float16*)carve((size_t)BH_ * T_ * HD_ * 2); // 8 MB
  _Float16* ehat  = (_Float16*)carve((size_t)H_ * T_ * HD_ * 2);  // 4 MB
  _Float16* ctxh  = (_Float16*)carve((size_t)BT_ * D_ * 2);       // 8 MB
  (void)ws_size; (void)in_sizes; (void)n_in; (void)out_size;

  const int nX = BT_ * D_;
  cvt_f16<<<(nX + 255) / 256, 256, 0, stream>>>(x, xh, nX);
  transpose16<<<(D_ * 3 * D_ + 255) / 256, 256, 0, stream>>>(Wqkv, wqkvT, D_,
                                                             3 * D_);
  transpose16<<<(D_ * D_ + 255) / 256, 256, 0, stream>>>(Wout, woutT, D_, D_);
  build_ehat<<<(H_ * T_ * HD_ + 255) / 256, 256, 0, stream>>>(ehat);

  qkv_gemm<<<dim3(3 * D_ / 64, BT_ / 256), 256, 0, stream>>>(xh, wqkvT, qh, kh,
                                                             vt);
  attn<<<dim3(T_ / 16, BH_), 32, 0, stream>>>(qh, kh, vt, ehat, score, ctxh);
  out_gemm<<<dim3(D_ / 64, BT_ / 256), 256, 0, stream>>>(ctxh, woutT, bout,
                                                         out);
}
